// MidmLMHeadModelWrapper_52312701665462
// MI455X (gfx1250) — compile-verified, matrix operand order checked by reading, not run on hardware
//
#include <hip/hip_runtime.h>
#include <hip/hip_fp16.h>
#include <math.h>

typedef __attribute__((ext_vector_type(16))) _Float16 v16h;
typedef __attribute__((ext_vector_type(8)))  _Float16 v8h;
typedef __attribute__((ext_vector_type(4)))  _Float16 v4h;
typedef __attribute__((ext_vector_type(8)))  float    v8f;

// Model dims (fixed by the reference)
constexpr int kS   = 1024;
constexpr int kD   = 2048;
constexpr int kH   = 16;
constexpr int kHD  = 128;
constexpr int kF   = 8192;
constexpr int kV   = 32000;
constexpr int kL   = 2;
constexpr int kROT = 64;

#define GF_BIAS 1
#define GF_RES  2
#define GF_GELU 4

// ---------------------------------------------------------------------------
// Embedding gather: x[s, :] = wte[ids[s], :]
// ---------------------------------------------------------------------------
__global__ void embed_kernel(const int* __restrict__ ids,
                             const float* __restrict__ wte,
                             float* __restrict__ x) {
  const int s = blockIdx.x;
  const long base = (long)ids[s] * kD;
  for (int d = threadIdx.x; d < kD; d += blockDim.x)
    x[s * kD + d] = wte[base + d];
}

// ---------------------------------------------------------------------------
// LayerNorm with (w+1) scale as in ln1p(); one block per row, 256 threads
// ---------------------------------------------------------------------------
__global__ __launch_bounds__(256) void ln_kernel(const float* __restrict__ in,
                                                 const float* __restrict__ w,
                                                 const float* __restrict__ b,
                                                 float* __restrict__ out) {
  const int row = blockIdx.x, tid = threadIdx.x;
  const float* x = in + row * kD;
  __shared__ float s1[256];
  __shared__ float s2[256];
  float sum = 0.f, sq = 0.f;
  for (int d = tid; d < kD; d += 256) {
    float v = x[d];
    sum += v;
    sq  += v * v;
  }
  s1[tid] = sum; s2[tid] = sq;
  __syncthreads();
  for (int s = 128; s > 0; s >>= 1) {
    if (tid < s) { s1[tid] += s1[tid + s]; s2[tid] += s2[tid + s]; }
    __syncthreads();
  }
  const float mu  = s1[0] * (1.0f / kD);
  const float var = s2[0] * (1.0f / kD) - mu * mu;
  const float rs  = rsqrtf(var + 1e-5f);
  for (int d = tid; d < kD; d += 256)
    out[row * kD + d] = (x[d] - mu) * rs * (w[d] + 1.0f) + b[d];
}

// ---------------------------------------------------------------------------
// WMMA GEMM: C[M,N] = A[M,K] @ B[K,N] (+bias)(+GELU)(+residual)
// fp32 in/out, f16 operands, f32 accumulation.
//
// Block = 256 threads = 8 waves. Block tile 128x128, BK = 32.
// DOUBLE-BUFFERED LDS staging (f16, padded stride 40 halves = 80B):
//   As[p][row][k] : 128 x 32 row-major
//   Bs[p][col][k] : B tile stored TRANSPOSED so a fragment is two
//                   contiguous ds_load_b128 per lane.
// Software pipeline per K-step:
//   issue global b128 loads of tile t+1 -> regs   (8 loads in flight)
//   ds_load fragments of tile t + 8 x v_wmma      (hides HBM latency)
//   convert regs -> LDS buffer p^1
//   one barrier; flip p
// Waves arranged 4 (rows) x 2 (cols): each wave computes 32x64 =
// 2 A-frags x 4 B-frags -> 8 v_wmma_f32_16x16x32_f16 per K-step.
//
// Fragment element mapping (ISA 7.12.2, wave32, 16-bit operands):
//   frag[e] <- K(e) = (e<8 ? e : e+8) + 8*(lane>=16); row/col = lane&15
// => frag = concat( LDS[r][8*h .. 8*h+7], LDS[r][16+8*h .. 16+8*h+7] )
//
// Requires M%128==0, N%128==0, K%32==0 (true for every call here).
// ---------------------------------------------------------------------------
__global__ __launch_bounds__(256) void gemm_wmma_kernel(
    const float* __restrict__ A, const float* __restrict__ B,
    const float* __restrict__ bias, const float* __restrict__ res,
    float* __restrict__ C, int M, int N, int K, int flags) {
  constexpr int LDT = 40;  // padded LDS row stride in halves (80 bytes)
  __shared__ __align__(16) _Float16 As[2][128][LDT];
  __shared__ __align__(16) _Float16 Bs[2][128][LDT];

  const int tid   = threadIdx.x;
  const int lane  = tid & 31;
  const int wave  = tid >> 5;
  const int hhalf = lane >> 4;       // 0 or 1
  const int idx   = lane & 15;

  const int blockRow = blockIdx.y * 128;
  const int blockCol = blockIdx.x * 128;
  const int wm = wave >> 1;          // 0..3 -> 32-row strip
  const int wn = wave & 1;           // 0..1 -> 64-col strip
  const int rowW = wm * 32;
  const int colW = wn * 64;

  // Staging maps
  const int a_kq  = tid & 7;         // 8 float4 quads along K
  const int a_row = tid >> 3;        // 0..31, step 32 (4 iters)
  const int b_nq  = tid & 31;        // 32 float4 quads along N
  const int b_k   = tid >> 5;        // 0..7, step 8 (4 iters)

  v8f acc[2][4];
#pragma unroll
  for (int i = 0; i < 2; ++i)
#pragma unroll
    for (int j = 0; j < 4; ++j) acc[i][j] = (v8f){};

  float4 fa[4], fb[4];

  // Batched global loads of one K-tile into registers (8 b128 in flight).
  auto load_tile = [&](int k0) {
#pragma unroll
    for (int it = 0; it < 4; ++it)
      fa[it] = *(const float4*)&A[(long)(blockRow + a_row + it * 32) * K + k0 + a_kq * 4];
#pragma unroll
    for (int it = 0; it < 4; ++it)
      fb[it] = *(const float4*)&B[(long)(k0 + b_k + it * 8) * N + blockCol + b_nq * 4];
  };
  // Convert f32->f16 and store into LDS buffer p.
  auto store_tile = [&](int p) {
#pragma unroll
    for (int it = 0; it < 4; ++it) {
      v4h pk;
      pk[0] = (_Float16)fa[it].x; pk[1] = (_Float16)fa[it].y;
      pk[2] = (_Float16)fa[it].z; pk[3] = (_Float16)fa[it].w;
      *(v4h*)&As[p][a_row + it * 32][a_kq * 4] = pk;
    }
#pragma unroll
    for (int it = 0; it < 4; ++it) {
      const int kk = b_k + it * 8;
      Bs[p][b_nq * 4 + 0][kk] = (_Float16)fb[it].x;
      Bs[p][b_nq * 4 + 1][kk] = (_Float16)fb[it].y;
      Bs[p][b_nq * 4 + 2][kk] = (_Float16)fb[it].z;
      Bs[p][b_nq * 4 + 3][kk] = (_Float16)fb[it].w;
    }
  };

  // Prologue: stage tile 0 into buffer 0.
  load_tile(0);
  store_tile(0);
  __syncthreads();

  int p = 0;
  for (int k0 = 0; k0 < K; k0 += 32) {
    const bool has_next = (k0 + 32) < K;
    if (has_next) load_tile(k0 + 32);   // global loads in flight during compute

    // ---- fragments of current tile from LDS: two b128 per frag per lane ----
    v16h af[2], bf[4];
#pragma unroll
    for (int rm = 0; rm < 2; ++rm) {
      const int r = rowW + rm * 16 + idx;
      const v8h lo = *(const v8h*)&As[p][r][8 * hhalf];
      const v8h hi = *(const v8h*)&As[p][r][16 + 8 * hhalf];
      af[rm] = __builtin_shufflevector(lo, hi, 0, 1, 2, 3, 4, 5, 6, 7,
                                       8, 9, 10, 11, 12, 13, 14, 15);
    }
#pragma unroll
    for (int cn = 0; cn < 4; ++cn) {
      const int c = colW + cn * 16 + idx;
      const v8h lo = *(const v8h*)&Bs[p][c][8 * hhalf];
      const v8h hi = *(const v8h*)&Bs[p][c][16 + 8 * hhalf];
      bf[cn] = __builtin_shufflevector(lo, hi, 0, 1, 2, 3, 4, 5, 6, 7,
                                       8, 9, 10, 11, 12, 13, 14, 15);
    }

    // ---- 8 WMMAs per K-step ----
#pragma unroll
    for (int rm = 0; rm < 2; ++rm)
#pragma unroll
      for (int cn = 0; cn < 4; ++cn)
        acc[rm][cn] = __builtin_amdgcn_wmma_f32_16x16x32_f16(
            false, af[rm], false, bf[cn], (short)0, acc[rm][cn], false, false);

    // ---- write next tile into the other buffer, single barrier ----
    if (has_next) store_tile(p ^ 1);
    __syncthreads();
    p ^= 1;
  }

  // ---- epilogue ----
#pragma unroll
  for (int rm = 0; rm < 2; ++rm) {
#pragma unroll
    for (int cn = 0; cn < 4; ++cn) {
#pragma unroll
      for (int r = 0; r < 8; ++r) {
        const int row = blockRow + rowW + rm * 16 + hhalf * 8 + r;
        const int col = blockCol + colW + cn * 16 + idx;
        float v = acc[rm][cn][r];
        if (flags & GF_BIAS) v += bias[col];
        if (flags & GF_GELU) v = 0.5f * v * (1.0f + erff(v * 0.70710678118654752f));
        if (flags & GF_RES)  v += res[(long)row * N + col];
        C[(long)row * N + col] = v;
      }
    }
  }
}

// ---------------------------------------------------------------------------
// RoPE applied in-place to q (z=0) and k (z=1) halves of qkv.
// Rotary span = first 64 dims of each 128-dim head; pairs (j, j+32).
// ---------------------------------------------------------------------------
__global__ void rope_kernel(float* __restrict__ qkv, const int* __restrict__ pos) {
  const int s = blockIdx.x, h = blockIdx.y, z = blockIdx.z;
  const int j = threadIdx.x;                 // 0..31
  const float p = (float)pos[s];
  // inv_freq[j] = 10000^(-2j/64)
  const float invf = __expf(-((float)(2 * j) / (float)kROT) * 9.2103403719761836f);
  const float f = p * invf;
  const float c = __cosf(f), sn = __sinf(f);
  float* t = qkv + (long)s * 3 * kD + z * kD + h * kHD;
  const float x1 = t[j];
  const float x2 = t[j + 32];
  t[j]      = x1 * c - x2 * sn;
  t[j + 32] = x2 * c + x1 * sn;
}

// ---------------------------------------------------------------------------
// Attention for one (query, head): scores -> softmax -> probs @ V
// scores = (q.k / sqrt(HD)) + (1 - mask)*(-1e4)*lscale   [== (s/l + m)*l]
// Block of 128 threads (4 waves). ~5 KB LDS.
// ---------------------------------------------------------------------------
__global__ __launch_bounds__(128) void attn_kernel(const float* __restrict__ qkv,
                                                   const float* __restrict__ mask,
                                                   float* __restrict__ o,
                                                   float lscale) {
  const int q = blockIdx.x, h = blockIdx.y, tid = threadIdx.x;
  const int srow = 3 * kD;
  __shared__ float qv[kHD];
  __shared__ float sc[kS];
  __shared__ float red[128];

  qv[tid] = qkv[(long)q * srow + h * kHD + tid];
  __syncthreads();

  const float inv_sqrt_hd = 0.08838834764831845f; // 1/sqrt(128)
  for (int kk = tid; kk < kS; kk += 128) {
    const float* kp = qkv + (long)kk * srow + kD + h * kHD;
    float acc = 0.f;
#pragma unroll 8
    for (int t = 0; t < kHD; ++t) acc += qv[t] * kp[t];
    const float m = mask[(long)q * kS + kk];
    sc[kk] = acc * inv_sqrt_hd + (1.0f - m) * (-10000.0f) * lscale;
  }
  __syncthreads();

  // max reduction
  float lmax = -1e30f;
  for (int kk = tid; kk < kS; kk += 128) lmax = fmaxf(lmax, sc[kk]);
  red[tid] = lmax;
  __syncthreads();
  for (int s = 64; s > 0; s >>= 1) {
    if (tid < s) red[tid] = fmaxf(red[tid], red[tid + s]);
    __syncthreads();
  }
  const float gmax = red[0];
  __syncthreads();

  // exp + sum reduction
  float lsum = 0.f;
  for (int kk = tid; kk < kS; kk += 128) {
    const float e = __expf(sc[kk] - gmax);
    sc[kk] = e;
    lsum += e;
  }
  red[tid] = lsum;
  __syncthreads();
  for (int s = 64; s > 0; s >>= 1) {
    if (tid < s) red[tid] += red[tid + s];
    __syncthreads();
  }
  const float inv = 1.0f / red[0];

  // o[tid] = sum_k probs[k] * V[k][tid]  (coalesced across threads)
  float accv = 0.f;
  for (int kk = 0; kk < kS; ++kk)
    accv += sc[kk] * qkv[(long)kk * srow + 2 * kD + h * kHD + tid];
  o[(long)q * kD + h * kHD + tid] = accv * inv;
}

// ---------------------------------------------------------------------------
// Host-side orchestration
// ---------------------------------------------------------------------------
extern "C" void kernel_launch(void* const* d_in, const int* in_sizes, int n_in,
                              void* d_out, int out_size, void* d_ws, size_t ws_size,
                              hipStream_t stream) {
  (void)in_sizes; (void)n_in; (void)out_size; (void)ws_size;

  const int*   ids   = (const int*)  d_in[0];
  const float* mask  = (const float*)d_in[1];
  const int*   cpos  = (const int*)  d_in[2];
  // d_in[3] batch_position (=-1), d_in[4] query_idx: full logits output
  const float* wte   = (const float*)d_in[5];
  const float* ln1w  = (const float*)d_in[6];
  const float* ln1b  = (const float*)d_in[7];
  const float* attnw = (const float*)d_in[8];
  const float* attnb = (const float*)d_in[9];
  const float* projw = (const float*)d_in[10];
  const float* projb = (const float*)d_in[11];
  const float* ln2w  = (const float*)d_in[12];
  const float* ln2b  = (const float*)d_in[13];
  const float* fcw   = (const float*)d_in[14];
  const float* fcb   = (const float*)d_in[15];
  const float* fcpw  = (const float*)d_in[16];
  const float* fcpb  = (const float*)d_in[17];
  const float* lnfw  = (const float*)d_in[18];
  const float* lnfb  = (const float*)d_in[19];
  const float* lmw   = (const float*)d_in[20];

  float* ws  = (float*)d_ws;
  float* x   = ws;                        // S*D
  float* hn  = x   + (size_t)kS * kD;     // S*D
  float* qkv = hn  + (size_t)kS * kD;     // S*3D
  float* o   = qkv + (size_t)kS * 3 * kD; // S*D
  float* mlp = o   + (size_t)kS * kD;     // S*F
  float* out = (float*)d_out;

  embed_kernel<<<kS, 256, 0, stream>>>(ids, wte, x);

  for (int li = 0; li < kL; ++li) {
    const float lscale = (float)li + 1.0f;

    // x -> hn -> qkv
    ln_kernel<<<kS, 256, 0, stream>>>(x, ln1w + li * kD, ln1b + li * kD, hn);
    gemm_wmma_kernel<<<dim3(3 * kD / 128, kS / 128), 256, 0, stream>>>(
        hn, attnw + (size_t)li * kD * 3 * kD, attnb + (size_t)li * 3 * kD,
        nullptr, qkv, kS, 3 * kD, kD, GF_BIAS);

    rope_kernel<<<dim3(kS, kH, 2), 32, 0, stream>>>(qkv, cpos);
    attn_kernel<<<dim3(kS, kH), 128, 0, stream>>>(qkv, mask, o, lscale);

    // x = x + o @ pw + pb
    gemm_wmma_kernel<<<dim3(kD / 128, kS / 128), 256, 0, stream>>>(
        o, projw + (size_t)li * kD * kD, projb + (size_t)li * kD,
        x, x, kS, kD, kD, GF_BIAS | GF_RES);

    // MLP
    ln_kernel<<<kS, 256, 0, stream>>>(x, ln2w + li * kD, ln2b + li * kD, hn);
    gemm_wmma_kernel<<<dim3(kF / 128, kS / 128), 256, 0, stream>>>(
        hn, fcw + (size_t)li * kD * kF, fcb + (size_t)li * kF,
        nullptr, mlp, kS, kF, kD, GF_BIAS | GF_GELU);
    gemm_wmma_kernel<<<dim3(kD / 128, kS / 128), 256, 0, stream>>>(
        mlp, fcpw + (size_t)li * kF * kD, fcpb + (size_t)li * kD,
        x, x, kS, kD, kF, GF_BIAS | GF_RES);
  }

  // Final LN + LM head
  ln_kernel<<<kS, 256, 0, stream>>>(x, lnfw, lnfb, hn);
  gemm_wmma_kernel<<<dim3(kV / 128, kS / 128), 256, 0, stream>>>(
      hn, lmw, nullptr, nullptr, out, kS, kV, kD, 0);
}
